// KANLayer_85005992722824
// MI455X (gfx1250) — matrix operand order repaired and checked by reading, not compiled, not run on hardware
//
#include <hip/hip_runtime.h>
#include <stdint.h>

#ifndef __has_builtin
#define __has_builtin(x) 0
#endif

typedef __attribute__((ext_vector_type(16))) _Float16     v16h;
typedef __attribute__((ext_vector_type(2)))  _Float16     h2;
typedef __attribute__((ext_vector_type(8)))  float        v8f;
typedef __attribute__((ext_vector_type(4)))  unsigned int u32x4;
typedef __attribute__((ext_vector_type(8)))  int          i32x8;
typedef __attribute__((ext_vector_type(4)))  int          i32x4;

#define BATCH 8192
#define FIN   256
#define KPTS  32
#define OOUT  64

#define FCHUNK      4                               // features staged per LDS chunk
#define CHUNK_HALF  (FCHUNK * 4 * 32 * 16)          // 8192 halves
#define CHUNK_BYTES (CHUNK_HALF * 2)                // 16 KB
#define NCHUNK      (FIN / FCHUNK)                  // 64

#if __has_builtin(__builtin_amdgcn_tensor_load_to_lds) && \
    __has_builtin(__builtin_amdgcn_s_wait_tensorcnt) &&   \
    __has_builtin(__builtin_amdgcn_groupstaticsize)
#define USE_TDM 1
#else
#define USE_TDM 0
#endif

// ---------------------------------------------------------------------------
// Prep: f32 kan_weight [F][K][O] -> f16 B pre-swizzled to WMMA B layout.
// dst halves indexed [f][ntile][lane][slot]:  K = 16*(lane/16)+slot,
//                                             N = 16*ntile + lane%16
// ---------------------------------------------------------------------------
__global__ __launch_bounds__(256) void kan_swizzle_b(const float* __restrict__ w,
                                                     _Float16* __restrict__ bsw) {
  int idx  = blockIdx.x * 256 + threadIdx.x;   // 0 .. F*K*O-1  (524288)
  int slot = idx & 15;
  int lane = (idx >> 4) & 31;
  int nt   = (idx >> 9) & 3;
  int f    = idx >> 11;
  int k = (lane >> 4) * 16 + slot;
  int n = nt * 16 + (lane & 15);
  bsw[idx] = (_Float16)w[(f * KPTS + k) * OOUT + n];
}

// ---------------------------------------------------------------------------
// Stage one 16KB B-chunk (4 features) into LDS buffer `buf`.
// TDM path: one 2D descriptor (4096 dwords x 1 row), issued by wave 0 only.
// ---------------------------------------------------------------------------
__device__ __forceinline__ void stage_chunk(const _Float16* __restrict__ bsw,
                                            char* smem, int chunk, int buf,
                                            int tid, int wave) {
#if USE_TDM
  if (wave == 0) {
    uint64_t ga  = (uint64_t)(uintptr_t)(bsw + (size_t)chunk * CHUNK_HALF);
    unsigned lds = __builtin_amdgcn_groupstaticsize() + (unsigned)buf * CHUNK_BYTES;
    u32x4 g0 = {};
    g0[0] = 1u;                                        // count=1, user descriptor
    g0[1] = lds;                                       // lds_addr
    g0[2] = (unsigned)(ga & 0xFFFFFFFFu);              // global_addr[31:0]
    g0[3] = (unsigned)((ga >> 32) & 0x01FFFFFFu)       // global_addr[56:32]
          | (2u << 30);                                // type = 2 ("image")
    i32x8 g1 = {};
    g1[0] = (2 << 16);                                 // data_size = 4 bytes
    g1[1] = (int)(4096u << 16);                        // tensor_dim0 = 4096 dwords
    g1[2] = (int)(1u << 16);                           // tensor_dim1 = 1
    g1[3] = (int)(4096u << 16);                        // tile_dim0   = 4096
    g1[4] = 1;                                         // tile_dim1   = 1
    g1[5] = 4096;                                      // tensor_dim0_stride
    i32x4 g2 = {};
    i32x4 g3 = {};
#if defined(__clang_major__) && (__clang_major__ >= 23)
    i32x8 g4 = {};
    __builtin_amdgcn_tensor_load_to_lds(g0, g1, g2, g3, g4, 0);
#else
    __builtin_amdgcn_tensor_load_to_lds(g0, g1, g2, g3, 0);
#endif
  }
#else
  // Fallback: cooperative copy, 64B per thread (4 x b128).
  const float4* src = (const float4*)(bsw + (size_t)chunk * CHUNK_HALF);
  float4* dst = (float4*)(smem + (size_t)buf * CHUNK_BYTES);
#pragma unroll
  for (int k = 0; k < 4; ++k) dst[tid + k * 256] = src[tid + k * 256];
#endif
}

__device__ __forceinline__ void wait_stage(int wave) {
#if USE_TDM
  if (wave == 0) __builtin_amdgcn_s_wait_tensorcnt(0);
#endif
}

// ---------------------------------------------------------------------------
// Main kernel: 8 waves/block; wave w owns batch rows [b0, b0+16), all O=64.
// A-build per feature in packed f16 (exact branchless select, 2 slots/op).
// ---------------------------------------------------------------------------
__global__ __launch_bounds__(256) void kan_wmma(const float* __restrict__ x,
                                                const _Float16* __restrict__ bsw,
                                                float* __restrict__ out) {
  extern __shared__ char smem[];                 // 2 * 16 KB double buffer
  const int tid  = threadIdx.x;
  const int wave = tid >> 5;
  const int lane = tid & 31;
  const int row  = lane & 15;                    // A/C row M handled by lane
  const int hi   = lane >> 4;                    // lane half selects K/M groups
  const int b0   = blockIdx.x * 128 + wave * 16;
  const int kbase = hi * 16;                     // A layout K base per lane half

  // Hoisted packed K constants: pair i holds K(2i), K(2i+1),
  // K(j) = kbase + j   (exact small ints in f16)
  h2 kf2[8];
#pragma unroll
  for (int i = 0; i < 8; ++i) {
    int j0 = 2 * i;
    int k0 = kbase + j0;
    kf2[i][0] = (_Float16)(float)k0;
    kf2[i][1] = (_Float16)(float)(k0 + 1);
  }
  const h2 one2  = {(_Float16)1.0f, (_Float16)1.0f};
  const h2 zero2 = {(_Float16)0.0f, (_Float16)0.0f};

  v8f acc0 = {}, acc1 = {}, acc2 = {}, acc3 = {};

  stage_chunk(bsw, smem, 0, 0, tid, wave);

  const float* xrow = x + (size_t)(b0 + row) * FIN;

  for (int fc = 0; fc < NCHUNK; ++fc) {
    const int buf = fc & 1;
    wait_stage(wave);
    __syncthreads();                             // chunk `buf` ready for all waves
    if (fc + 1 < NCHUNK)
      stage_chunk(bsw, smem, fc + 1, buf ^ 1, tid, wave);   // prefetch next

    const float4 xv = *(const float4*)(xrow + fc * 4);      // 4 features of x
    const char* bbase = smem + (size_t)buf * CHUNK_BYTES;

#pragma unroll
    for (int fi = 0; fi < FCHUNK; ++fi) {
      const float xf = (fi == 0) ? xv.x : (fi == 1) ? xv.y : (fi == 2) ? xv.z : xv.w;
      // xs in control-point coords; clamp lower index like the reference
      float xs = (xf + 2.0f) * (31.0f / 4.0f);
      float lo = floorf(xs);
      lo = fminf(fmaxf(lo, 0.0f), 30.0f);
      const float t = xs - lo;

      // Broadcast packed-f16 scalars for this feature
      const _Float16 lo_h = (_Float16)lo;
      const _Float16 bs_h = (_Float16)(1.0f - t);        // value at d==0
      const _Float16 tt_h = (_Float16)(2.0f * t - 1.0f); // slope: d=1 -> t
      const h2 lo2 = {lo_h, lo_h};
      const h2 bs2 = {bs_h, bs_h};
      const h2 tt2 = {tt_h, tt_h};

      // A slot j (K=kf): d = kf - lo;  w = [d∈{0,1}] * ((1-t) + d*(2t-1))
      //   [d∈{0,1}] = relu(1 - g*g),  g = d*d - d   (exact: g=0 iff d∈{0,1})
      v16h A;
#pragma unroll
      for (int i = 0; i < 8; ++i) {
        h2 d = kf2[i] - lo2;                                       // v_pk_add
        h2 g = __builtin_elementwise_fma(d, d, -d);                // v_pk_fma
        h2 s = __builtin_elementwise_fma(-g, g, one2);             // 1 - g^2
        s = __builtin_elementwise_max(s, zero2);                   // v_pk_max
        h2 m = __builtin_elementwise_fma(d, tt2, bs2);             // v_pk_fma
        h2 w = s * m;                                              // v_pk_mul
        A[2 * i]     = w[0];
        A[2 * i + 1] = w[1];
      }

      const char* bfeat = bbase + fi * 4096 + lane * 32;
      const v16h B0 = *(const v16h*)(bfeat +    0);
      const v16h B1 = *(const v16h*)(bfeat + 1024);
      const v16h B2 = *(const v16h*)(bfeat + 2048);
      const v16h B3 = *(const v16h*)(bfeat + 3072);

      acc0 = __builtin_amdgcn_wmma_f32_16x16x32_f16(false, A, false, B0, (short)0, acc0, false, false);
      acc1 = __builtin_amdgcn_wmma_f32_16x16x32_f16(false, A, false, B1, (short)0, acc1, false, false);
      acc2 = __builtin_amdgcn_wmma_f32_16x16x32_f16(false, A, false, B2, (short)0, acc2, false, false);
      acc3 = __builtin_amdgcn_wmma_f32_16x16x32_f16(false, A, false, B3, (short)0, acc3, false, false);
    }
    __syncthreads();                             // all reads of `buf` done
  }

  // C/D layout: VGPR r, lane L -> M = r + 8*(L>=16), N = L%16
  const int n0    = lane & 15;
  const int mBase = hi * 8;
#pragma unroll
  for (int r = 0; r < 8; ++r) {
    float* p = out + (size_t)(b0 + mBase + r) * OOUT + n0;
    p[0]  = acc0[r];
    p[16] = acc1[r];
    p[32] = acc2[r];
    p[48] = acc3[r];
  }
}

// ---------------------------------------------------------------------------
extern "C" void kernel_launch(void* const* d_in, const int* in_sizes, int n_in,
                              void* d_out, int out_size, void* d_ws, size_t ws_size,
                              hipStream_t stream) {
  (void)in_sizes; (void)n_in; (void)out_size; (void)ws_size;
  const float* x  = (const float*)d_in[0];           // [8192, 256]
  const float* kw = (const float*)d_in[1];           // [256, 32, 64]
  float* out      = (float*)d_out;                   // [8192, 64]
  _Float16* bsw   = (_Float16*)d_ws;                 // 1 MB swizzled f16 table

  kan_swizzle_b<<<(FIN * KPTS * OOUT) / 256, 256, 0, stream>>>(kw, bsw);
  kan_wmma<<<BATCH / 128, 256, 2 * CHUNK_BYTES, stream>>>(x, bsw, out);
}
